// SigFormer_44573170598757
// MI455X (gfx1250) — matrix operand order repaired and checked by reading, not compiled, not run on hardware
//
#include <hip/hip_runtime.h>
#include <hip/hip_bf16.h>
#include <math.h>

// ---------- CDNA5 types ----------
typedef __bf16 bf16;
typedef __attribute__((ext_vector_type(16))) __bf16 v16bf;
typedef __attribute__((ext_vector_type(8)))  __bf16 v8bf;
typedef __attribute__((ext_vector_type(8)))  float  v8f;

#define BATCH 4
#define SEQ   1024
#define DIM   512
#define HEADS 8
#define DH    64
#define MLPD  2048
#define NROWS (BATCH*SEQ)   // 4096

static __device__ __forceinline__ v16bf cat8(v8bf lo, v8bf hi) {
    return __builtin_shufflevector(lo, hi, 0,1,2,3,4,5,6,7,8,9,10,11,12,13,14,15);
}
static __device__ __forceinline__ v8f wmma_bf16(v16bf a, v16bf b, v8f c) {
    return __builtin_amdgcn_wmma_f32_16x16x32_bf16(false, a, false, b, (short)0, c, false, false);
}

// ---------- async Global->LDS (CDNA5, ASYNCcnt) with sync fallback ----------
#if defined(__has_builtin)
# if __has_builtin(__builtin_amdgcn_global_load_async_to_lds_b128)
#  define HAVE_ASYNC_LDS 1
# endif
#endif

static __device__ __forceinline__ void cp16_g2l(const bf16* g, bf16* l) {
#ifdef HAVE_ASYNC_LDS
    typedef int v4i_ __attribute__((vector_size(16)));
    typedef __attribute__((address_space(1))) v4i_* gptr_t;   // pointer to AS1 v4i
    typedef __attribute__((address_space(3))) v4i_* lptr_t;   // pointer to AS3 v4i
    __builtin_amdgcn_global_load_async_to_lds_b128(
        (gptr_t)const_cast<bf16*>(g), (lptr_t)l, 0, 0);
#else
    *(uint4*)l = *(const uint4*)g;
#endif
}
static __device__ __forceinline__ void cp_g2l_wait() {
#ifdef HAVE_ASYNC_LDS
# if __has_builtin(__builtin_amdgcn_s_wait_asynccnt)
    __builtin_amdgcn_s_wait_asynccnt(0);
# else
    asm volatile("s_wait_asynccnt 0x0" ::: "memory");
# endif
#endif
}

// ---------- f32 -> bf16 cast ----------
__global__ void cast_f32_bf16(const float* __restrict__ in, bf16* __restrict__ out, int n) {
    int i = blockIdx.x * 256 + threadIdx.x;
    if (i < n) out[i] = (bf16)in[i];
}

// ---------- Generic bf16 GEMM: C[M,N] = A[M,K] @ B[K,N] (+bias, +gelu) ----------
// Block: 256 thr (8 waves). Tile: 128(M) x 64(N) x 32(K). Wave w -> rows [16w,16w+16), all 64 cols.
// Double-buffered LDS; A tile staged via async Global->LDS, B tile prefetched through registers.
__global__ __launch_bounds__(256) void gemm_bf16_k(
    const bf16* __restrict__ A, const bf16* __restrict__ Bm, void* __restrict__ C,
    const float* __restrict__ bias, int M, int N, int K, int outBf16, int doGelu)
{
    __shared__ bf16 sA[2][128 * 32];   // row-major [row][k]
    __shared__ bf16 sB[2][64 * 32];    // transposed [n][k]
    const int tid = threadIdx.x, lane = tid & 31, w = tid >> 5;
    const int bm = blockIdx.y * 128, bn = blockIdx.x * 64;
    const int r = lane & 15, hi = lane >> 4;

    v8f acc[4] = {};

    // A staging: each thread owns a 32B (16 bf16) chunk of the 128x32 tile
    const int arow = tid >> 1, aseg = tid & 1;
    const bf16* Asrc = A + (size_t)(bm + arow) * K + aseg * 16;
    const int aoff = arow * 32 + aseg * 16;
    // B staging: thread reads 8 contiguous bf16 of row (k0+bkk), scatters transposed
    const int bkk = tid >> 3, bnseg = tid & 7;
    const bf16* Bsrc = Bm + (size_t)bkk * N + bn + bnseg * 8;

    const int nk = K >> 5;

    // ---- prologue: stage tile 0 into buffer 0 ----
    cp16_g2l(Asrc,     &sA[0][aoff]);
    cp16_g2l(Asrc + 8, &sA[0][aoff + 8]);
    v8bf breg = *(const v8bf*)Bsrc;
    #pragma unroll
    for (int e = 0; e < 8; ++e) sB[0][(bnseg * 8 + e) * 32 + bkk] = breg[e];
    cp_g2l_wait();
    __syncthreads();

    for (int t = 0; t < nk; ++t) {
        const int cur = t & 1, nxt = cur ^ 1;
        const bool more = (t + 1 < nk);
        if (more) {
            const int k1 = (t + 1) << 5;
            cp16_g2l(Asrc + k1,     &sA[nxt][aoff]);       // async: overlaps WMMAs below
            cp16_g2l(Asrc + k1 + 8, &sA[nxt][aoff + 8]);
            breg = *(const v8bf*)(Bsrc + (size_t)k1 * N);  // global prefetch into regs
        }

        // A fragment: lane<16 -> K{0..7,16..23}; lane>=16 -> K{8..15,24..31}
        const bf16* ap = &sA[cur][(w * 16 + r) * 32 + (hi ? 8 : 0)];
        v16bf af = cat8(*(const v8bf*)ap, *(const v8bf*)(ap + 16));
        const int kb = hi ? 8 : 0;
        v16bf bfrag[4];
        #pragma unroll
        for (int j = 0; j < 4; ++j) {
            const bf16* bp = &sB[cur][(j * 16 + r) * 32 + kb];
            bfrag[j] = cat8(*(const v8bf*)bp, *(const v8bf*)(bp + 16));
        }
        #pragma unroll
        for (int j = 0; j < 4; ++j)
            acc[j] = wmma_bf16(af, bfrag[j], acc[j]);

        if (more) {
            #pragma unroll
            for (int e = 0; e < 8; ++e) sB[nxt][(bnseg * 8 + e) * 32 + bkk] = breg[e];
            cp_g2l_wait();
            __syncthreads();
        }
    }

    const int rowb = bm + w * 16 + hi * 8;
    #pragma unroll
    for (int j = 0; j < 4; ++j) {
        const int col = bn + j * 16 + r;
        const float bv = bias ? bias[col] : 0.0f;
        #pragma unroll
        for (int rr = 0; rr < 8; ++rr) {
            float v = acc[j][rr] + bv;
            if (doGelu) v = 0.5f * v * (1.0f + erff(v * 0.70710678118654752f));
            const size_t idx = (size_t)(rowb + rr) * N + col;
            if (outBf16) ((bf16*)C)[idx] = (bf16)v;
            else         ((float*)C)[idx] = v;
        }
    }
}

// ---------- Attention: one block (4 waves) per (b, h, 16-query tile) ----------
// LDS: scores 16x1024 f32 (64KB); bf16 P packed into upper half of each row slab; Vt 64x32 bf16 (4KB)
__global__ __launch_bounds__(128) void attn_k(
    const bf16* __restrict__ q, const bf16* __restrict__ k, const bf16* __restrict__ v,
    bf16* __restrict__ o)
{
    extern __shared__ char smem[];
    float* sc = (float*)smem;                        // 16*1024 f32
    bf16*  Vt = (bf16*)(smem + 16 * 1024 * 4);       // [n(dh)][kk]  64*32

    const int tid = threadIdx.x, lane = tid & 31, w = tid >> 5;
    const int qt = blockIdx.x, h = blockIdx.y, b = blockIdx.z;
    const int rb = b * SEQ, hc = h * DH;
    const int r = lane & 15, hi = lane >> 4;

    // q A-fragments for K-steps 0 and 1 (dh = 64 = 2x32)
    const bf16* qp = q + (size_t)(rb + qt * 16 + r) * DIM + hc + (hi ? 8 : 0);
    v16bf aq0 = cat8(*(const v8bf*)qp,        *(const v8bf*)(qp + 16));
    v16bf aq1 = cat8(*(const v8bf*)(qp + 32), *(const v8bf*)(qp + 48));

    // Phase 1: S = qk^T / 8 ; wave w covers key tiles [16w,16w+16)
    for (int t = 0; t < 16; ++t) {
        const int kt = w * 16 + t;
        const bf16* kp = k + (size_t)(rb + kt * 16 + r) * DIM + hc + (hi ? 8 : 0);
        v16bf b0 = cat8(*(const v8bf*)kp,        *(const v8bf*)(kp + 16));
        v16bf b1 = cat8(*(const v8bf*)(kp + 32), *(const v8bf*)(kp + 48));
        v8f s = {};
        s = wmma_bf16(aq0, b0, s);
        s = wmma_bf16(aq1, b1, s);
        #pragma unroll
        for (int rr = 0; rr < 8; ++rr)
            sc[(rr + hi * 8) * 1024 + kt * 16 + r] = s[rr] * 0.125f;
    }
    __syncthreads();

    // Phase 2: softmax rows 4w..4w+3; write bf16 P into bytes [2048,4096) of each row slab
    for (int rowi = 0; rowi < 4; ++rowi) {
        const int row = w * 4 + rowi;
        float vals[32];
        float m = -3.0e38f;
        #pragma unroll
        for (int i = 0; i < 32; ++i) { vals[i] = sc[row * 1024 + lane + i * 32]; m = fmaxf(m, vals[i]); }
        #pragma unroll
        for (int off = 16; off > 0; off >>= 1) m = fmaxf(m, __shfl_xor(m, off, 32));
        float s = 0.0f;
        #pragma unroll
        for (int i = 0; i < 32; ++i) { vals[i] = expf(vals[i] - m); s += vals[i]; }
        #pragma unroll
        for (int off = 16; off > 0; off >>= 1) s += __shfl_xor(s, off, 32);
        const float inv = 1.0f / s;
        bf16* Pr = (bf16*)(sc + row * 1024 + 512);
        #pragma unroll
        for (int i = 0; i < 32; ++i) Pr[lane + i * 32] = (bf16)(vals[i] * inv);
    }
    __syncthreads();

    // Phase 3: out = P @ V ; wave w owns dh columns [16w,16w+16)
    v8f oacc = {};
    for (int ks = 0; ks < 32; ++ks) {
        { // cooperative transpose of V tile (32 keys x 64 dh) -> Vt[n][kk]
            const int kk = tid >> 2, nseg = tid & 3;
            const bf16* vp = v + (size_t)(rb + ks * 32 + kk) * DIM + hc + nseg * 16;
            v8bf r0 = *(const v8bf*)vp;
            v8bf r1 = *(const v8bf*)(vp + 8);
            #pragma unroll
            for (int e = 0; e < 8; ++e) {
                Vt[(nseg * 16 + e) * 32 + kk]     = r0[e];
                Vt[(nseg * 16 + 8 + e) * 32 + kk] = r1[e];
            }
        }
        __syncthreads();
        const bf16* pp = (const bf16*)(sc + r * 1024 + 512) + ks * 32 + (hi ? 8 : 0);
        v16bf af = cat8(*(const v8bf*)pp, *(const v8bf*)(pp + 16));
        const bf16* bp = &Vt[(w * 16 + r) * 32 + (hi ? 8 : 0)];
        v16bf bf_ = cat8(*(const v8bf*)bp, *(const v8bf*)(bp + 16));
        oacc = wmma_bf16(af, bf_, oacc);
        __syncthreads();
    }
    #pragma unroll
    for (int rr = 0; rr < 8; ++rr) {
        const size_t row = (size_t)rb + qt * 16 + rr + hi * 8;
        o[row * DIM + hc + w * 16 + r] = (bf16)oacc[rr];
    }
}

// ---------- LayerNorm(a + res) -> z_f32, z_bf16 ; wave per row, D=512 ----------
__global__ __launch_bounds__(256) void ln_k(
    const float* __restrict__ a, const float* __restrict__ res,
    const float* __restrict__ g, const float* __restrict__ bb,
    float* __restrict__ z32, bf16* __restrict__ zb)
{
    const int lane = threadIdx.x & 31, w = threadIdx.x >> 5;
    const int row = blockIdx.x * 8 + w;
    const float* pa = a + (size_t)row * DIM;
    const float* pr = res + (size_t)row * DIM;
    float vals[16];
    float s = 0.0f;
    #pragma unroll
    for (int i = 0; i < 16; ++i) { const int c = lane + i * 32; vals[i] = pa[c] + pr[c]; s += vals[i]; }
    #pragma unroll
    for (int off = 16; off > 0; off >>= 1) s += __shfl_xor(s, off, 32);
    const float mu = s * (1.0f / DIM);
    float qv = 0.0f;
    #pragma unroll
    for (int i = 0; i < 16; ++i) { const float d = vals[i] - mu; qv += d * d; }
    #pragma unroll
    for (int off = 16; off > 0; off >>= 1) qv += __shfl_xor(qv, off, 32);
    const float rs = rsqrtf(qv * (1.0f / DIM) + 1e-5f);
    #pragma unroll
    for (int i = 0; i < 16; ++i) {
        const int c = lane + i * 32;
        const float ov = (vals[i] - mu) * rs * g[c] + bb[c];
        z32[(size_t)row * DIM + c] = ov;
        zb[(size_t)row * DIM + c] = (bf16)ov;
    }
}

// ---------- (B,T,C) -> (B,C,T) transpose ----------
__global__ void transpose_k(const float* __restrict__ in, float* __restrict__ out) {
    __shared__ float tile[32][33];
    const int b = blockIdx.z;
    const int t0 = blockIdx.x * 32, c0 = blockIdx.y * 32;
    for (int j = threadIdx.y; j < 32; j += 8)
        tile[j][threadIdx.x] = in[((size_t)b * SEQ + t0 + j) * DIM + c0 + threadIdx.x];
    __syncthreads();
    for (int j = threadIdx.y; j < 32; j += 8)
        out[((size_t)b * DIM + c0 + j) * SEQ + t0 + threadIdx.x] = tile[threadIdx.x][j];
}

// ---------- dual-barrier weights: w[b][k][t] (K=31) and wsum[b][t] ----------
__global__ void barrier_w_k(const float* __restrict__ bar, float* __restrict__ wbuf, float* __restrict__ wsum) {
    const int id = blockIdx.x * 256 + threadIdx.x;
    if (id >= BATCH * SEQ) return;
    const int b = id >> 10, t = id & 1023;
    float bu[31];
    #pragma unroll
    for (int kk = 0; kk < 31; ++kk) {
        const int idx = t + kk - 15;
        bu[kk] = (idx >= 0 && idx < SEQ) ? bar[b * SEQ + idx] : 20.0f;
    }
    float wv[31];
    wv[15] = 1.0f;                      // exp(-alpha*0)
    float acc = bu[15];
    for (int j = 14; j >= 0; --j) { acc += bu[j]; wv[j] = expf(-0.2f * acc); }          // left
    acc = bu[15];
    for (int j = 0; j < 15; ++j)  { acc += bu[16 + j]; wv[16 + j] = expf(-0.2f * acc); } // right
    float s = 0.0f;
    #pragma unroll
    for (int kk = 0; kk < 31; ++kk) { s += wv[kk]; wbuf[((size_t)b * 31 + kk) * SEQ + t] = wv[kk]; }
    wsum[b * SEQ + t] = s;
}

// ---------- local barrier pooling: out (B,C,T) ----------
__global__ void pool_k(const float* __restrict__ zc, const float* __restrict__ wbuf,
                       const float* __restrict__ wsum, float* __restrict__ out) {
    const int id = blockIdx.x * 256 + threadIdx.x;   // B*C*T = 2^21
    const int t = id & 1023, c = (id >> 10) & 511, b = id >> 19;
    float acc = 0.0f;
    #pragma unroll
    for (int kk = 0; kk < 31; ++kk) {
        const int tt = t + kk - 15;
        const float xv = (tt >= 0 && tt < SEQ) ? zc[((size_t)b * DIM + c) * SEQ + tt] : 0.0f;
        acc += xv * wbuf[((size_t)b * 31 + kk) * SEQ + t];
    }
    out[id] = acc / (wsum[b * SEQ + t] + 4.5399929762484854e-05f);
}

// =====================================================================
extern "C" void kernel_launch(void* const* d_in, const int* in_sizes, int n_in,
                              void* d_out, int out_size, void* d_ws, size_t ws_size,
                              hipStream_t stream) {
    (void)in_sizes; (void)n_in; (void)out_size; (void)ws_size;
    const float* x   = (const float*)d_in[0];
    const float* y   = (const float*)d_in[1];
    const float* bar = (const float*)d_in[2];
    const float* Wq  = (const float*)d_in[3];
    const float* Wk  = (const float*)d_in[4];
    const float* Wv  = (const float*)d_in[5];
    const float* Wo  = (const float*)d_in[6];
    const float* ln1g = (const float*)d_in[7];
    const float* ln1b = (const float*)d_in[8];
    const float* W1  = (const float*)d_in[9];
    const float* b1  = (const float*)d_in[10];
    const float* W2  = (const float*)d_in[11];
    const float* b2  = (const float*)d_in[12];
    const float* ln2g = (const float*)d_in[13];
    const float* ln2b = (const float*)d_in[14];
    float* out = (float*)d_out;

    // ---- workspace layout (256B aligned slices) ----
    char* ws = (char*)d_ws;
    size_t off = 0;
    auto alloc = [&](size_t bytes) { size_t o = off; off = (off + bytes + 255) & ~(size_t)255; return o; };
    const size_t EL = (size_t)NROWS * DIM;            // 4096*512
    bf16* xbf  = (bf16*)(ws + alloc(EL * 2));
    bf16* ybf  = (bf16*)(ws + alloc(EL * 2));
    bf16* wqb  = (bf16*)(ws + alloc((size_t)2 * DIM * DIM * 2));
    bf16* wkb  = (bf16*)(ws + alloc((size_t)2 * DIM * DIM * 2));
    bf16* wvb  = (bf16*)(ws + alloc((size_t)2 * DIM * DIM * 2));
    bf16* wob  = (bf16*)(ws + alloc((size_t)2 * DIM * DIM * 2));
    bf16* w1b  = (bf16*)(ws + alloc((size_t)2 * DIM * MLPD * 2));
    bf16* w2b  = (bf16*)(ws + alloc((size_t)2 * MLPD * DIM * 2));
    bf16* qbf  = (bf16*)(ws + alloc(EL * 2));
    bf16* kbf  = (bf16*)(ws + alloc(EL * 2));
    bf16* vbf  = (bf16*)(ws + alloc(EL * 2));
    bf16* atb  = (bf16*)(ws + alloc(EL * 2));          // attention output (bf16)
    float* gbuf = (float*)(ws + alloc(EL * 4));        // attn-proj out / mlp2 out (f32)
    float* zA   = (float*)(ws + alloc(EL * 4));
    float* zB   = (float*)(ws + alloc(EL * 4));
    bf16* zbf   = (bf16*)(ws + alloc(EL * 2));
    bf16* hbf   = (bf16*)(ws + alloc((size_t)NROWS * MLPD * 2));
    float* zc   = (float*)(ws + alloc(EL * 4));
    float* wbuf = (float*)(ws + alloc((size_t)BATCH * 31 * SEQ * 4));
    float* wsum = (float*)(ws + alloc((size_t)BATCH * SEQ * 4));

    // ---- casts to bf16 ----
    auto cast = [&](const float* src, bf16* dst, int n) {
        cast_f32_bf16<<<(n + 255) / 256, 256, 0, stream>>>(src, dst, n);
    };
    cast(x, xbf, (int)EL);
    cast(y, ybf, (int)EL);
    cast(Wq, wqb, 2 * DIM * DIM);
    cast(Wk, wkb, 2 * DIM * DIM);
    cast(Wv, wvb, 2 * DIM * DIM);
    cast(Wo, wob, 2 * DIM * DIM);
    cast(W1, w1b, 2 * DIM * MLPD);
    cast(W2, w2b, 2 * MLPD * DIM);

    const dim3 gProj(DIM / 64, NROWS / 128);   // (8,32)
    const dim3 gMlp1(MLPD / 64, NROWS / 128);  // (32,32)
    const size_t attnLds = 16 * 1024 * 4 + 64 * 32 * 2;   // 69632 B

    for (int d = 0; d < 2; ++d) {
        const bf16* wq_d = wqb + (size_t)d * DIM * DIM;
        const bf16* wk_d = wkb + (size_t)d * DIM * DIM;
        const bf16* wv_d = wvb + (size_t)d * DIM * DIM;
        const bf16* wo_d = wob + (size_t)d * DIM * DIM;
        const bf16* w1_d = w1b + (size_t)d * DIM * MLPD;
        const bf16* w2_d = w2b + (size_t)d * MLPD * DIM;

        // q/k/v projections (queries from ORIGINAL x each depth, per reference)
        gemm_bf16_k<<<gProj, 256, 0, stream>>>(xbf, wq_d, qbf, nullptr, NROWS, DIM, DIM, 1, 0);
        gemm_bf16_k<<<gProj, 256, 0, stream>>>(ybf, wk_d, kbf, nullptr, NROWS, DIM, DIM, 1, 0);
        gemm_bf16_k<<<gProj, 256, 0, stream>>>(ybf, wv_d, vbf, nullptr, NROWS, DIM, DIM, 1, 0);

        attn_k<<<dim3(SEQ / 16, HEADS, BATCH), 128, attnLds, stream>>>(qbf, kbf, vbf, atb);

        // output projection (f32), then z = LN(a + x)
        gemm_bf16_k<<<gProj, 256, 0, stream>>>(atb, wo_d, gbuf, nullptr, NROWS, DIM, DIM, 0, 0);
        ln_k<<<NROWS / 8, 256, 0, stream>>>(gbuf, x, ln1g + d * DIM, ln1b + d * DIM, zA, zbf);

        // MLP: h = gelu(z@W1+b1) ; f = h@W2+b2 ; z = LN(f + z)
        gemm_bf16_k<<<gMlp1, 256, 0, stream>>>(zbf, w1_d, hbf, b1 + d * MLPD, NROWS, MLPD, DIM, 1, 1);
        gemm_bf16_k<<<gProj, 256, 0, stream>>>(hbf, w2_d, gbuf, b2 + d * DIM, NROWS, DIM, MLPD, 0, 0);
        ln_k<<<NROWS / 8, 256, 0, stream>>>(gbuf, zA, ln2g + d * DIM, ln2b + d * DIM, zB, zbf);
    }

    // ---- local barrier pooling ----
    transpose_k<<<dim3(SEQ / 32, DIM / 32, BATCH), dim3(32, 8), 0, stream>>>(zB, zc);
    barrier_w_k<<<(BATCH * SEQ) / 256, 256, 0, stream>>>(bar, wbuf, wsum);
    pool_k<<<(BATCH * DIM * SEQ) / 256, 256, 0, stream>>>(zc, wbuf, wsum, out);
}
